// RNN_75273596829743
// MI455X (gfx1250) — matrix operand order repaired and checked by reading, not compiled, not run on hardware
//
#include <hip/hip_runtime.h>
#include <hip/hip_bf16.h>
#include <math.h>

// ---------------------------------------------------------------------------
// Problem constants (from reference): IN=1024, H=20, L=4, B=1, T=32768
// ---------------------------------------------------------------------------
#define T_LEN   32768
#define IN_D    1024
#define HS      20
#define GATES   80          // 4*H
#define NLAYER  4

typedef __attribute__((ext_vector_type(16))) _Float16 v16h;
typedef __attribute__((ext_vector_type(8)))  _Float16 v8h;
typedef __attribute__((ext_vector_type(8)))  float    v8f;

__device__ __forceinline__ float fast_sigmoid(float x) {
    return 1.0f / (1.0f + __expf(-x));
}
__device__ __forceinline__ float fast_tanh(float x) {
    const float e = __expf(-2.0f * x);
    return (1.0f - e) / (1.0f + e);
}

// ---------------------------------------------------------------------------
// Kernel 1: proj[t][g] = x[t,:] . W_ih0[g,:] + b_ih0[g] + b_hh0[g]
// WMMA f16 GEMM: M = T (tiles of 16), N = 80 (5 tiles of 16, one per wave),
// K = 1024 (chunks of 32). Bandwidth-bound: x read once as float4.
// ---------------------------------------------------------------------------
__global__ __launch_bounds__(160) void proj_gemm(
    const float* __restrict__ x,      // [T, IN]
    const float* __restrict__ Wih0,   // [80, IN]
    const float* __restrict__ bih0,   // [80]
    const float* __restrict__ bhh0,   // [80]
    float* __restrict__ proj)         // [T, 80]
{
    __shared__ __align__(16) _Float16 sA[16 * 32];
    __shared__ __align__(16) _Float16 sB[5][16 * 32];

    const int tid    = threadIdx.x;
    const int wave   = tid >> 5;       // 0..4  -> N tile
    const int lane   = tid & 31;
    const int rowSel = lane & 15;      // M row (A) / N col (B) for this lane
    const int hi     = lane >> 4;      // lane half selects K sub-range
    const int m0     = blockIdx.x * 16;

    v8f acc = {};

    for (int kc = 0; kc < IN_D; kc += 32) {
        // ---- stage A (shared by all 5 waves): 128 threads x 4 floats ----
        if (tid < 128) {
            const int r = tid >> 3;            // 0..15
            const int c = (tid & 7) * 4;       // 0,4,...,28
            const float4 v =
                *(const float4*)(x + (size_t)(m0 + r) * IN_D + kc + c);
            _Float16* dst = &sA[r * 32 + c];
            dst[0] = (_Float16)v.x; dst[1] = (_Float16)v.y;
            dst[2] = (_Float16)v.z; dst[3] = (_Float16)v.w;
        }
        // ---- stage B (per wave): lane stages half a W row (16 floats) ----
        {
            const int n  = rowSel;
            const int ck = hi * 16;
            const float* src = Wih0 + (size_t)(wave * 16 + n) * IN_D + kc + ck;
            _Float16* dst = &sB[wave][n * 32 + ck];
            #pragma unroll
            for (int q = 0; q < 4; ++q) {
                const float4 v = *(const float4*)(src + q * 4);
                dst[q * 4 + 0] = (_Float16)v.x; dst[q * 4 + 1] = (_Float16)v.y;
                dst[q * 4 + 2] = (_Float16)v.z; dst[q * 4 + 3] = (_Float16)v.w;
            }
        }
        __syncthreads();

        // A fragment: 16-bit A 16x32 layout (ISA 7.12.2):
        //   lanes 0-15 : M=lane,    K = 0..7  then 16..23
        //   lanes 16-31: M=lane-16, K = 8..15 then 24..31
        const int ka0 = hi * 8;
        const v8h a0 = *(const v8h*)&sA[rowSel * 32 + ka0];
        const v8h a1 = *(const v8h*)&sA[rowSel * 32 + ka0 + 16];
        v16h a;
        #pragma unroll
        for (int e = 0; e < 8; ++e) { a[e] = a0[e]; a[8 + e] = a1[e]; }

        // B fragment: 16-bit B 32x16 layout (matches A):
        //   lanes 0-15 : N=lane,    K = 0..7  then 16..23
        //   lanes 16-31: N=lane-16, K = 8..15 then 24..31
        const int kb0 = hi * 8;
        const v8h b0 = *(const v8h*)&sB[wave][rowSel * 32 + kb0];
        const v8h b1 = *(const v8h*)&sB[wave][rowSel * 32 + kb0 + 16];
        v16h b;
        #pragma unroll
        for (int e = 0; e < 8; ++e) { b[e] = b0[e]; b[8 + e] = b1[e]; }

        acc = __builtin_amdgcn_wmma_f32_16x16x32_f16(
            /*neg_a=*/false, a, /*neg_b=*/false, b,
            /*c_mod=*/(short)0, acc, /*reuse_a=*/false, /*reuse_b=*/false);

        __syncthreads();
    }

    // epilogue: C layout VGPR r -> M = r + 8*hi, N = lane&15
    const int g    = wave * 16 + rowSel;
    const float bz = bih0[g] + bhh0[g];
    #pragma unroll
    for (int r8 = 0; r8 < 8; ++r8) {
        const int M = r8 + hi * 8;
        proj[(size_t)(m0 + M) * GATES + g] = acc[r8] + bz;
    }
}

// ---------------------------------------------------------------------------
// Kernel 2: 4-layer LSTM scan, software-pipelined across 4 waves.
// Wave l = layer l, processes step t = k - l; h vectors move through LDS
// double buffers; one workgroup barrier per pipeline tick.
//
// Gate GEMVs spread over ALL 32 lanes: lane L owns gates {L, L+32, L+64},
// each a 40-term dot over xin = [h_in(20) | h_prev(20)] with register-
// resident weights. Pre-activations are exchanged via per-layer LDS (same
// wave -> in-order LDS, no barrier; explicit s_wait_dscnt + wave_barrier
// stop compiler reordering). Lane j<20 then owns unit j's c/h update.
// ---------------------------------------------------------------------------
__global__ __launch_bounds__(128) void lstm_scan(
    const float* __restrict__ proj,    // [T, 80] (layer-0 proj, biases folded)
    const float* __restrict__ Whh0,    // [80, 20]
    const float* __restrict__ Wih_r,   // [3, 80, 20]
    const float* __restrict__ Whh_r,   // [3, 80, 20]
    const float* __restrict__ bih_r,   // [3, 80]
    const float* __restrict__ bhh_r,   // [3, 80]
    const float* __restrict__ lin_w,   // [1, 20]
    const float* __restrict__ lin_b,   // [1]
    float* __restrict__ out)           // [1]
{
    __shared__ float sWih[NLAYER][GATES][HS];  // layer 0 slot zeroed
    __shared__ float sWhh[NLAYER][GATES][HS];
    __shared__ float sBias[NLAYER][GATES];
    __shared__ float sG[NLAYER][GATES];        // gate pre-activations (per tick)
    __shared__ float hbuf[NLAYER][2][HS];      // double-buffered by t&1

    const int tid  = threadIdx.x;
    const int wave = tid >> 5;                 // layer index
    const int lane = tid & 31;
    const int j    = lane;                     // hidden unit (lanes < 20)

    // ---- cooperative weight/bias staging into LDS ----
    for (int i = tid; i < GATES * HS; i += 128) {
        const int g = i / HS, k = i % HS;
        sWhh[0][g][k] = Whh0[i];
        sWih[0][g][k] = 0.0f;
        #pragma unroll
        for (int l = 1; l < NLAYER; ++l) {
            sWih[l][g][k] = Wih_r[(size_t)(l - 1) * GATES * HS + i];
            sWhh[l][g][k] = Whh_r[(size_t)(l - 1) * GATES * HS + i];
        }
    }
    for (int g = tid; g < GATES; g += 128) {
        sBias[0][g] = 0.0f;   // folded into proj by the GEMM kernel
        #pragma unroll
        for (int l = 1; l < NLAYER; ++l)
            sBias[l][g] = bih_r[(l - 1) * GATES + g] + bhh_r[(l - 1) * GATES + g];
    }
    __syncthreads();

    // ---- hoist this lane's gate weights into registers (fully unrolled) ----
    // wv[gi][0..19]  = W_ih[g][:]   (input half)
    // wv[gi][20..39] = W_hh[g][:]   (recurrent half)
    float wv[3][2 * HS], bz[3];
    #pragma unroll
    for (int gi = 0; gi < 3; ++gi) {
        const int g = lane + gi * 32;
        if (g < GATES) {
            bz[gi] = sBias[wave][g];
            #pragma unroll
            for (int k = 0; k < HS; ++k) {
                wv[gi][k]      = sWih[wave][g][k];
                wv[gi][HS + k] = sWhh[wave][g][k];
            }
        } else {
            bz[gi] = 0.0f;
            #pragma unroll
            for (int k = 0; k < 2 * HS; ++k) wv[gi][k] = 0.0f;
        }
    }

    float c = 0.0f;

    // ---- prefetch proj row 0 (wave 0 only) ----
    float pn[3] = {0.f, 0.f, 0.f};
    if (wave == 0) {
        #pragma unroll
        for (int gi = 0; gi < 3; ++gi) {
            const int g = lane + gi * 32;
            if (g < GATES) pn[gi] = proj[g];
        }
    }

    for (int kk = 0; kk < T_LEN + NLAYER - 1; ++kk) {
        const int t = kk - wave;

        float pc[3];
        #pragma unroll
        for (int gi = 0; gi < 3; ++gi) pc[gi] = pn[gi];
        // prefetch next step's proj row; latency hidden behind this tick
        if (wave == 0 && (kk + 1) < T_LEN) {
            #pragma unroll
            for (int gi = 0; gi < 3; ++gi) {
                const int g = lane + gi * 32;
                if (g < GATES) pn[gi] = proj[(size_t)(kk + 1) * GATES + g];
            }
        }

        if (t >= 0 && t < T_LEN) {
            // gather inputs (replicated across lanes, LDS broadcast reads)
            float xin[2 * HS];
            #pragma unroll
            for (int k = 0; k < HS; ++k) {
                xin[k]      = (wave == 0) ? 0.0f : hbuf[wave - 1][t & 1][k];
                xin[HS + k] = (t == 0)    ? 0.0f : hbuf[wave][(t & 1) ^ 1][k];
            }
            // gate pre-activations: up to 3 independent 40-term FMA chains
            float accv[3];
            #pragma unroll
            for (int gi = 0; gi < 3; ++gi)
                accv[gi] = (wave == 0) ? pc[gi] : bz[gi];
            #pragma unroll
            for (int k = 0; k < 2 * HS; ++k) {
                #pragma unroll
                for (int gi = 0; gi < 3; ++gi)
                    accv[gi] = fmaf(xin[k], wv[gi][k], accv[gi]);
            }
            #pragma unroll
            for (int gi = 0; gi < 3; ++gi) {
                const int g = lane + gi * 32;
                if (g < GATES) sG[wave][g] = accv[gi];
            }
            // same-wave LDS exchange: hardware keeps DS ops in order; just
            // stop the compiler from hoisting the loads above the stores.
            asm volatile("s_wait_dscnt 0x0" ::: "memory");
            __builtin_amdgcn_wave_barrier();

            if (j < HS) {
                const float ig = fast_sigmoid(sG[wave][j]);
                const float fg = fast_sigmoid(sG[wave][HS + j]);
                const float gg = fast_tanh   (sG[wave][2 * HS + j]);
                const float og = fast_sigmoid(sG[wave][3 * HS + j]);
                c = fg * c + ig * gg;
                hbuf[wave][t & 1][j] = og * fast_tanh(c);
            }
        }
        __syncthreads();   // pipeline tick: publish h[t], protect t&1 buffers
    }

    // ---- head: sigmoid(h3[T-1] . lin_w + lin_b) ----
    if (tid == 0) {
        float d = lin_b[0];
        #pragma unroll
        for (int k = 0; k < HS; ++k)
            d += hbuf[NLAYER - 1][(T_LEN - 1) & 1][k] * lin_w[k];
        out[0] = fast_sigmoid(d);
    }
}

// ---------------------------------------------------------------------------
// Launch: inputs in setup_inputs() order.
//   0:x 1:W_ih0 2:W_hh0 3:b_ih0 4:b_hh0 5:W_ih_r 6:W_hh_r 7:b_ih_r 8:b_hh_r
//   9:lin_w 10:lin_b
// d_ws holds proj: 32768*80 floats = 10.5 MB.
// ---------------------------------------------------------------------------
extern "C" void kernel_launch(void* const* d_in, const int* in_sizes, int n_in,
                              void* d_out, int out_size, void* d_ws, size_t ws_size,
                              hipStream_t stream) {
    const float* x     = (const float*)d_in[0];
    const float* Wih0  = (const float*)d_in[1];
    const float* Whh0  = (const float*)d_in[2];
    const float* bih0  = (const float*)d_in[3];
    const float* bhh0  = (const float*)d_in[4];
    const float* Wih_r = (const float*)d_in[5];
    const float* Whh_r = (const float*)d_in[6];
    const float* bih_r = (const float*)d_in[7];
    const float* bhh_r = (const float*)d_in[8];
    const float* lin_w = (const float*)d_in[9];
    const float* lin_b = (const float*)d_in[10];

    float* proj = (float*)d_ws;   // [T_LEN, GATES]

    proj_gemm<<<T_LEN / 16, 160, 0, stream>>>(x, Wih0, bih0, bhh0, proj);
    lstm_scan<<<1, 128, 0, stream>>>(proj, Whh0, Wih_r, Whh_r, bih_r, bhh_r,
                                     lin_w, lin_b, (float*)d_out);
}
